// DecoderLSTMWithAttention_15109694948162
// MI455X (gfx1250) — compile-verified, matrix operand order checked by reading, not run on hardware
//
#include <hip/hip_runtime.h>
#include <hip/hip_bf16.h>

// ---------------------------------------------------------------------------
// DecoderLSTMWithAttention for MI455X (gfx1250, wave32, WMMA).
//
// Algebra: the 4-layer attention MLP is linear and the h-dependent part of the
// score is constant over the softmax axis -> attention weights / ctx are
// h-independent, computed ONCE. The [x_t, ctx] @ W_ih^T term is precomputed
// for all 27 steps in one GEMM. Per-step work is only the h-dependent gates
// GEMM (128x512x2048) and the logits GEMM (128x512x30000), both on
// v_wmma_f32_16x16x32_bf16 with fp32 accumulation.
//
// GEMM kernel: wave-level 16x16 tiles, NB-wide N blocking (NB divides the
// n-tile count exactly -> no bounds checks -> EXEC stays all-ones around
// WMMA, per ISA 7.12 requirement). Wave id goes through readfirstlane so the
// grid-stride loop is provably scalar (no saveexec around WMMA).
// ---------------------------------------------------------------------------

typedef unsigned short ushort_t;
typedef __attribute__((ext_vector_type(16))) __bf16 v16bf;
typedef __attribute__((ext_vector_type(8)))  float  v8f;

#define B_SZ  128
#define H_SZ  512
#define S_SZ  80
#define T_SZ  28
#define TS_SZ 27      // scan steps = T-1
#define V_SZ  30000
#define D_SZ  512

__device__ __forceinline__ ushort_t f2bf(float f) {
  unsigned int u = __builtin_bit_cast(unsigned int, f);
  unsigned int r = u + 0x7FFFu + ((u >> 16) & 1u);   // round-to-nearest-even
  return (ushort_t)(r >> 16);
}

// --------------------------- small utility kernels -------------------------

__global__ void f32_to_bf16_k(const float* __restrict__ in, ushort_t* __restrict__ out, long n) {
  long i = (long)blockIdx.x * blockDim.x + threadIdx.x;
  long s = (long)gridDim.x * blockDim.x;
  for (; i < n; i += s) out[i] = f2bf(in[i]);
}

__global__ void fill_f32_k(float* __restrict__ p, float v, long n) {
  long i = (long)blockIdx.x * blockDim.x + threadIdx.x;
  long s = (long)gridDim.x * blockDim.x;
  for (; i < n; i += s) p[i] = v;
}

__global__ void vadd_k(const float* __restrict__ a, const float* __restrict__ b,
                       float* __restrict__ o, int n) {
  int i = blockIdx.x * blockDim.x + threadIdx.x;
  if (i < n) o[i] = a[i] + b[i];
}

// vout[j] = sum_k W[k*cols + j] * vin[k]   (i.e. W^T @ vin), j < outn
__global__ void matT_vec_k(const float* __restrict__ W, int rows, int cols,
                           const float* __restrict__ vin, float* __restrict__ vout, int outn) {
  int j = blockIdx.x * blockDim.x + threadIdx.x;
  if (j >= outn) return;
  float s = 0.f;
  for (int k = 0; k < rows; ++k) s += W[(long)k * cols + j] * vin[k];
  vout[j] = s;
}

// ------------------- one-time attention (h-independent) --------------------
__global__ void attention_ctx_k(const float* __restrict__ enc, const float* __restrict__ u,
                                float* __restrict__ ctxf) {
  __shared__ float sc[S_SZ];
  int b = blockIdx.x;
  const float* E = enc + (long)b * S_SZ * H_SZ;
  int tid = threadIdx.x, lane = tid & 31, wid = tid >> 5;   // 8 waves
  for (int s = wid; s < S_SZ; s += 8) {
    float p = 0.f;
    for (int h = lane; h < H_SZ; h += 32) p += E[s * H_SZ + h] * u[h];
    for (int o = 16; o; o >>= 1) p += __shfl_xor(p, o, 32);
    if (lane == 0) sc[s] = p;
  }
  __syncthreads();
  if (wid == 0) {   // softmax over S=80 in one wave
    float m = -3.4e38f;
    for (int s = lane; s < S_SZ; s += 32) m = fmaxf(m, sc[s]);
    for (int o = 16; o; o >>= 1) m = fmaxf(m, __shfl_xor(m, o, 32));
    float sum = 0.f;
    for (int s = lane; s < S_SZ; s += 32) { float e = __expf(sc[s] - m); sc[s] = e; sum += e; }
    for (int o = 16; o; o >>= 1) sum += __shfl_xor(sum, o, 32);
    float inv = 1.f / sum;
    for (int s = lane; s < S_SZ; s += 32) sc[s] *= inv;
  }
  __syncthreads();
  for (int h = tid; h < H_SZ; h += blockDim.x) {
    float acc = 0.f;
    for (int s = 0; s < S_SZ; ++s) acc += sc[s] * E[s * H_SZ + h];
    ctxf[(long)b * H_SZ + h] = acc;
  }
}

// li[t*B+b, 0:512] = embedding[targets[b, t]] ; li[.., 512:1024] = ctx[b]  (bf16)
__global__ void assemble_li_k(const float* __restrict__ embd, const int* __restrict__ tgt,
                              const float* __restrict__ ctxf, ushort_t* __restrict__ li) {
  long n = (long)TS_SZ * B_SZ * 1024;
  long i = (long)blockIdx.x * blockDim.x + threadIdx.x;
  long s = (long)gridDim.x * blockDim.x;
  for (; i < n; i += s) {
    long row = i >> 10; int col = (int)(i & 1023);
    int b = (int)(row & (B_SZ - 1)); int t = (int)(row >> 7);
    float v;
    if (col < D_SZ) {
      int tok = tgt[b * T_SZ + t];
      v = embd[(long)tok * D_SZ + col];
    } else {
      v = ctxf[(long)b * H_SZ + (col - D_SZ)];
    }
    li[i] = f2bf(v);
  }
}

// ------------------------------ WMMA GEMM ----------------------------------
// C[M,N] (f32) = A[M,K](bf16,rm) @ B[N,K](bf16,rm)^T (+ bias[N]) (+ Cadd[M,N])
// Fragment layout per ISA 7.12.2 (16-bit A 16x32): lane L holds row (L&15),
// two contiguous 8-elem K chunks at k+((L&16)>>1) and k+16+((L&16)>>1).
// B^T column n == row n of B (contiguous), same per-lane pattern.
__device__ __forceinline__ v16bf load_frag_ptr(const ushort_t* __restrict__ q) {
  union { v16bf v; uint4 x[2]; } u;
  u.x[0] = *(const uint4*)(q);
  u.x[1] = *(const uint4*)(q + 16);
  return u.v;
}

template <int NB, bool HAS_BIAS, bool HAS_CADD>
__global__ void wmma_gemm_nt_k(const ushort_t* __restrict__ A, int lda,
                               const ushort_t* __restrict__ B, int ldb,
                               float* __restrict__ C, long ldc,
                               const float* __restrict__ bias,
                               const float* __restrict__ Cadd, long ldca,
                               int M, int N, int K) {
  const int tiles_m = M >> 4;
  const int ngrp    = (N >> 4) / NB;       // must divide exactly (caller ensures)
  const long total  = (long)tiles_m * ngrp;
  // scalar wave id -> scalar grid-stride loop -> EXEC untouched around WMMA
  const int  wib    = __builtin_amdgcn_readfirstlane(threadIdx.x >> 5);
  const int  nwaves = gridDim.x * 8;       // blockDim.x == 256 always
  const int  lane   = threadIdx.x & 31;
  const int  lrow   = lane & 15;
  const int  lklo   = (lane & 16) >> 1;

  for (long g = (long)blockIdx.x * 8 + wib; g < total; g += nwaves) {
    const int tm = (int)(g / ngrp);
    const int gn = (int)(g % ngrp);
    const int m0 = tm << 4;
    const int n0 = gn * (NB << 4);

    const ushort_t* pa = A + (long)(m0 + lrow) * lda + lklo;
    const ushort_t* pb[NB];
#pragma unroll
    for (int j = 0; j < NB; ++j)
      pb[j] = B + (long)(n0 + (j << 4) + lrow) * ldb + lklo;

    v8f acc[NB];
#pragma unroll
    for (int j = 0; j < NB; ++j)
#pragma unroll
      for (int e = 0; e < 8; ++e) acc[j][e] = 0.f;

    for (int k = 0; k < K; k += 32) {
      v16bf a = load_frag_ptr(pa + k);
#pragma unroll
      for (int j = 0; j < NB; ++j) {
        v16bf b = load_frag_ptr(pb[j] + k);
        acc[j] = __builtin_amdgcn_wmma_f32_16x16x32_bf16(
            false, a, false, b, (short)0, acc[j], false, false);
      }
    }

    const int colbase = lane & 15;
    const int rofs    = (lane & 16) ? 8 : 0;
#pragma unroll
    for (int j = 0; j < NB; ++j) {
      const int col = n0 + (j << 4) + colbase;
      float bb = 0.f;
      if (HAS_BIAS) bb = bias[col];
#pragma unroll
      for (int r = 0; r < 8; ++r) {
        const long row = m0 + r + rofs;
        float v = acc[j][r] + bb;
        if (HAS_CADD) v += Cadd[row * ldca + col];
        C[row * ldc + col] = v;
      }
    }
  }
}

// ------------------------------ LSTM cell ----------------------------------
__global__ void lstm_cell_k(const float* __restrict__ gates, float* __restrict__ c,
                            ushort_t* __restrict__ hbf, int n) {
  int i = blockIdx.x * blockDim.x + threadIdx.x;
  if (i >= n) return;
  int b = i >> 9, j = i & (H_SZ - 1);
  const float* g = gates + (long)b * 4 * H_SZ;
  float ig = g[j], fg = g[H_SZ + j], gg = g[2 * H_SZ + j], og = g[3 * H_SZ + j];
  float si = 1.f / (1.f + __expf(-ig));
  float sf = 1.f / (1.f + __expf(-fg));
  float so = 1.f / (1.f + __expf(-og));
  float c2 = sf * c[i] + si * tanhf(gg);
  float h2 = so * tanhf(c2);
  c[i] = c2;
  hbf[i] = f2bf(h2);
}

// ------------------------------ argmax -------------------------------------
__global__ void argmax_rows_k(const float* __restrict__ L, long ldl, int ncols,
                              int* __restrict__ outp, int ostride) {
  __shared__ float sv[256];
  __shared__ int   si[256];
  int b = blockIdx.x, tid = threadIdx.x;
  const float* row = L + (long)b * ldl;
  float bv = -3.4e38f; int bi = 0;
  for (int j = tid; j < ncols; j += 256) {
    float v = row[j];
    if (v > bv) { bv = v; bi = j; }
  }
  sv[tid] = bv; si[tid] = bi;
  __syncthreads();
  for (int s = 128; s; s >>= 1) {
    if (tid < s) {
      float ov = sv[tid + s]; int oi = si[tid + s];
      if (ov > sv[tid] || (ov == sv[tid] && oi < si[tid])) { sv[tid] = ov; si[tid] = oi; }
    }
    __syncthreads();
  }
  if (tid == 0) outp[(long)b * ostride] = si[0];
}

// ------------------------------ host driver --------------------------------
extern "C" void kernel_launch(void* const* d_in, const int* in_sizes, int n_in,
                              void* d_out, int out_size, void* d_ws, size_t ws_size,
                              hipStream_t stream) {
  const float* h0   = (const float*)d_in[0];   // (1,B,H)
  const float* enc  = (const float*)d_in[1];   // (B,S,H)
  const int*   tgt  = (const int*)  d_in[2];   // (B,T)
  const float* embd = (const float*)d_in[3];   // (V,D)
  const float* W1   = (const float*)d_in[4];
  const float* W2   = (const float*)d_in[6];
  const float* W3   = (const float*)d_in[8];
  const float* W4   = (const float*)d_in[10];
  const float* watt = (const float*)d_in[12];  // (1,H)
  const float* Wih  = (const float*)d_in[13];  // (4H, H+D)
  const float* Whh  = (const float*)d_in[14];  // (4H, H)
  const float* bih  = (const float*)d_in[15];
  const float* bhh  = (const float*)d_in[16];
  const float* Wout = (const float*)d_in[17];  // (V, H)
  const float* bout = (const float*)d_in[18];
  (void)in_sizes; (void)n_in; (void)out_size; (void)ws_size;

  float* out  = (float*)d_out;                              // (B, TS, V) logits
  int*   pred = (int*)(out + (size_t)B_SZ * TS_SZ * V_SZ);  // (B, TS) argmax

  char* wsb = (char*)d_ws; size_t off = 0;
  auto alloc = [&](size_t bytes) -> void* {
    void* p = wsb + off;
    off += (bytes + 255) & ~(size_t)255;
    return p;
  };
  float*    tA    = (float*)alloc(512 * 4);
  float*    tB    = (float*)alloc(512 * 4);
  float*    u_e   = (float*)alloc(512 * 4);
  float*    biasc = (float*)alloc(2048 * 4);
  float*    ctxf  = (float*)alloc((size_t)B_SZ * H_SZ * 4);
  float*    cst   = (float*)alloc((size_t)B_SZ * H_SZ * 4);
  float*    gates = (float*)alloc((size_t)B_SZ * 2048 * 4);
  float*    P     = (float*)alloc((size_t)TS_SZ * B_SZ * 2048 * 4);
  ushort_t* hbf   = (ushort_t*)alloc((size_t)B_SZ * H_SZ * 2);
  ushort_t* wihb  = (ushort_t*)alloc((size_t)2048 * 1024 * 2);
  ushort_t* whhb  = (ushort_t*)alloc((size_t)2048 * 512 * 2);
  ushort_t* woutb = (ushort_t*)alloc((size_t)V_SZ * 512 * 2);
  ushort_t* lib   = (ushort_t*)alloc((size_t)TS_SZ * B_SZ * 1024 * 2);

  // one-time: weight conversions + state init
  f32_to_bf16_k<<<256, 256, 0, stream>>>(Wih,  wihb,  (long)2048 * 1024);
  f32_to_bf16_k<<<256, 256, 0, stream>>>(Whh,  whhb,  (long)2048 * 512);
  f32_to_bf16_k<<<1024, 256, 0, stream>>>(Wout, woutb, (long)V_SZ * 512);
  f32_to_bf16_k<<<64, 256, 0, stream>>>(h0,   hbf,   (long)B_SZ * H_SZ);
  fill_f32_k<<<64, 256, 0, stream>>>(cst, 0.f, (long)B_SZ * H_SZ);
  vadd_k<<<8, 256, 0, stream>>>(bih, bhh, biasc, 2048);

  // one-time: collapsed linear attention vector u_e = W1^T W2^T W3^T W4^T w
  matT_vec_k<<<2, 256, 0, stream>>>(W4, 512, 512,  watt, tA, 512);
  matT_vec_k<<<2, 256, 0, stream>>>(W3, 512, 512,  tA,   tB, 512);
  matT_vec_k<<<2, 256, 0, stream>>>(W2, 512, 512,  tB,   tA, 512);
  matT_vec_k<<<2, 256, 0, stream>>>(W1, 512, 1024, tA,   u_e, 512);  // enc half

  // one-time: softmax over enc scores (h-invariant) + ctx
  attention_ctx_k<<<B_SZ, 256, 0, stream>>>(enc, u_e, ctxf);

  // one-time: P[t,b,:] = [x_t, ctx] @ Wih^T + (b_ih + b_hh)  for all 27 steps
  assemble_li_k<<<1024, 256, 0, stream>>>(embd, tgt, ctxf, lib);
  // M=3456 (216 m-tiles), N=2048 (128 n-tiles, NB=4 -> 32 groups): 6912 waves
  wmma_gemm_nt_k<4, true, false><<<864, 256, 0, stream>>>(
      lib, 1024, wihb, 1024, P, 2048, biasc, nullptr, 0,
      TS_SZ * B_SZ, 2048, 1024);

  // sequential scan: only h-dependent GEMMs remain per step
  for (int t = 0; t < TS_SZ; ++t) {
    // gates = P[t] + h @ Whh^T   (M=128 -> 8 m-tiles, N=2048: 256 waves)
    wmma_gemm_nt_k<4, false, true><<<32, 256, 0, stream>>>(
        hbf, 512, whhb, 512, gates, 2048, nullptr,
        P + (size_t)t * B_SZ * 2048, 2048, B_SZ, 2048, 512);
    lstm_cell_k<<<256, 256, 0, stream>>>(gates, cst, hbf, B_SZ * H_SZ);
    // logits = h2 @ Wout^T + b_out (N=30000: 1875 n-tiles, NB=5 -> 375 groups,
    // 8*375 = 3000 waves -> 375 blocks)
    wmma_gemm_nt_k<5, true, false><<<375, 256, 0, stream>>>(
        hbf, 512, woutb, 512, out + (size_t)t * V_SZ, (long)TS_SZ * V_SZ,
        bout, nullptr, 0, B_SZ, V_SZ, 512);
    argmax_rows_k<<<B_SZ, 256, 0, stream>>>(out + (size_t)t * V_SZ, (long)TS_SZ * V_SZ,
                                            V_SZ, pred + t, TS_SZ);
  }
}
